// LightningQuantumGCN_40905268527217
// MI455X (gfx1250) — compile-verified, matrix operand order checked by reading, not run on hardware
//
#include <hip/hip_runtime.h>
#include <math.h>

typedef __attribute__((ext_vector_type(16))) _Float16 v16h;
typedef __attribute__((ext_vector_type(8)))  float    v8f;

#define NN 50000
#define EE 800000
#define ETOT (EE + NN)
#define DD 64
#define MAXPH 0.78539816339f   // pi/4

__device__ __forceinline__ v8f wmma16(v16h a, v16h b, v8f c) {
  // D = A(16x32 f16) * B(32x16 f16) + C(16x16 f32)
  return __builtin_amdgcn_wmma_f32_16x16x32_f16(false, a, false, b, (short)0, c, false, false);
}

// K index pattern for 16-bit 16x32 A/B fragments: element e, half = lane>>4
__device__ __forceinline__ int kpat(int e, int half) {
  return (e & 7) + ((e >> 3) << 4) + (half << 3);
}

// ---------------------------------------------------------------------------
// Kernel 1: x_real = LayerNorm(x); h = x_real @ Wr (WMMA); mag = sqrt(h^2+eps)
// one wave per 16-row tile, 8 waves per block
// ---------------------------------------------------------------------------
__global__ void __launch_bounds__(256)
ln_h_mag_kernel(const float* __restrict__ x, const float* __restrict__ W,
                float* __restrict__ xr, float* __restrict__ h,
                float* __restrict__ mg) {
  __shared__ float tileS[8][16][65];
  __shared__ float statS[8][16][2];
  int wave = threadIdx.x >> 5, lane = threadIdx.x & 31;
  int half = lane >> 4, l15 = lane & 15;
  int rowTile = blockIdx.x * 8 + wave;
  bool act = (rowTile * 16) < NN;
  int row0 = rowTile * 16;
  if (act) {
#pragma unroll
    for (int t = 0; t < 32; ++t) {
      int li = t * 32 + lane; int r = li >> 6, c = li & 63;
      tileS[wave][r][c] = x[(size_t)(row0 + r) * DD + c];
    }
  }
  __syncthreads();
  if (act && lane < 16) {
    float s1 = 0.f, s2 = 0.f;
#pragma unroll
    for (int c = 0; c < 64; ++c) { float v = tileS[wave][lane][c]; s1 += v; s2 += v * v; }
    float m = s1 * (1.f / 64.f);
    float var = s2 * (1.f / 64.f) - m * m;
    statS[wave][lane][0] = m;
    statS[wave][lane][1] = rsqrtf(var + 1e-5f);
  }
  __syncthreads();
  if (!act) return;
#pragma unroll
  for (int t = 0; t < 32; ++t) {  // coalesced x_real store
    int li = t * 32 + lane; int r = li >> 6, c = li & 63;
    float v = (tileS[wave][r][c] - statS[wave][r][0]) * statS[wave][r][1];
    xr[(size_t)(row0 + r) * DD + c] = v;
  }
  float m = statS[wave][l15][0], rs = statS[wave][l15][1];
  v16h a0, a1;
#pragma unroll
  for (int e = 0; e < 16; ++e) {
    int K = kpat(e, half);
    a0[e] = (_Float16)((tileS[wave][l15][K] - m) * rs);
    a1[e] = (_Float16)((tileS[wave][l15][K + 32] - m) * rs);
  }
#pragma unroll
  for (int ct = 0; ct < 4; ++ct) {
    int col = ct * 16 + l15;
    v16h b0, b1;
#pragma unroll
    for (int e = 0; e < 16; ++e) {
      int K = kpat(e, half);
      b0[e] = (_Float16)W[K * DD + col];         // W[k][n], row-major [64][64]
      b1[e] = (_Float16)W[(K + 32) * DD + col];
    }
    v8f acc = {};
    acc = wmma16(a0, b0, acc);
    acc = wmma16(a1, b1, acc);
#pragma unroll
    for (int r = 0; r < 8; ++r) {
      size_t idx = (size_t)(row0 + r + 8 * half) * DD + col;
      float v = acc[r];
      h[idx] = v;
      mg[idx] = sqrtf(v * v + 1e-8f);
    }
  }
}

// ---------------------------------------------------------------------------
// Kernel 2: P = h @ Ur^T, Q = h @ Ui^T (WMMA)
// ---------------------------------------------------------------------------
__global__ void __launch_bounds__(256)
pq_kernel(const float* __restrict__ h, const float* __restrict__ Ur,
          const float* __restrict__ Ui, float* __restrict__ P,
          float* __restrict__ Q) {
  int wave = threadIdx.x >> 5, lane = threadIdx.x & 31;
  int half = lane >> 4, l15 = lane & 15;
  int rowTile = blockIdx.x * 8 + wave;
  if (rowTile * 16 >= NN) return;
  int row0 = rowTile * 16;
  v16h a0, a1;
#pragma unroll
  for (int e = 0; e < 16; ++e) {
    int K = kpat(e, half);
    a0[e] = (_Float16)h[(size_t)(row0 + l15) * DD + K];
    a1[e] = (_Float16)h[(size_t)(row0 + l15) * DD + K + 32];
  }
#pragma unroll
  for (int ct = 0; ct < 4; ++ct) {
    int col = ct * 16 + l15;
    v16h br0, br1, bi0, bi1;
#pragma unroll
    for (int e = 0; e < 16; ++e) {
      int K = kpat(e, half);
      br0[e] = (_Float16)Ur[col * DD + K];        // (Ur^T)[K][col] = Ur[col][K]
      br1[e] = (_Float16)Ur[col * DD + K + 32];
      bi0[e] = (_Float16)Ui[col * DD + K];
      bi1[e] = (_Float16)Ui[col * DD + K + 32];
    }
    v8f pa = {}, qa = {};
    pa = wmma16(a0, br0, pa);
    pa = wmma16(a1, br1, pa);
    qa = wmma16(a0, bi0, qa);
    qa = wmma16(a1, bi1, qa);
#pragma unroll
    for (int r = 0; r < 8; ++r) {
      size_t idx = (size_t)(row0 + r + 8 * half) * DD + col;
      P[idx] = pa[r];
      Q[idx] = qa[r];
    }
  }
}

// ---------------------------------------------------------------------------
// Kernel 3: per-edge MLP (WMMA, 16 edges per wave) + rotate + scatter-add
// ---------------------------------------------------------------------------
__global__ void __launch_bounds__(256)
edge_kernel(const int* __restrict__ src, const int* __restrict__ dst,
            const float* __restrict__ ew,
            const float* __restrict__ mg, const float* __restrict__ P,
            const float* __restrict__ Q,
            const float* __restrict__ lng, const float* __restrict__ lnb,
            const float* __restrict__ W1, const float* __restrict__ b1,
            const float* __restrict__ W2, const float* __restrict__ b2,
            float* __restrict__ outR, float* __restrict__ outI,
            int nIter, int nTiles) {
  __shared__ _Float16 featS[8][16][128];
  __shared__ float statS[8][16][2];
  __shared__ float csS[8][16][2];
  __shared__ int   sdS[8][16][2];
  __shared__ float ewS[8][16];
  int wave = threadIdx.x >> 5, lane = threadIdx.x & 31;
  int half = lane >> 4, l15 = lane & 15;
  // preload W1 [128,32] B fragments: [col-tile][k-chunk]
  v16h bw[2][4];
#pragma unroll
  for (int ct = 0; ct < 2; ++ct)
#pragma unroll
    for (int ck = 0; ck < 4; ++ck) {
      int col = ct * 16 + l15;
#pragma unroll
      for (int e = 0; e < 16; ++e) {
        int K = ck * 32 + kpat(e, half);
        bw[ct][ck][e] = (_Float16)W1[K * 32 + col];
      }
    }
  float w2lo = W2[l15], w2hi = W2[16 + l15];
  float b1lo = b1[l15], b1hi = b1[16 + l15];
  float b2v = b2[0];

  for (int it = 0; it < nIter; ++it) {
    int tile = (it * gridDim.x + blockIdx.x) * 8 + wave;
    bool act = tile < nTiles;
    if (act && lane < 16) {
      int ge = tile * 16 + lane;
      int s_, d_; float w;
      if (ge < EE) { s_ = src[ge]; d_ = dst[ge]; w = ew[ge]; }
      else         { s_ = d_ = ge - EE; w = 1.f; }           // self loops
      sdS[wave][lane][0] = s_; sdS[wave][lane][1] = d_;
      ewS[wave][lane] = w;
    }
    __syncthreads();
    if (act) {
#pragma unroll 1
      for (int e = 0; e < 16; ++e) {      // build feat + per-edge LN stats
        int s_ = sdS[wave][e][0], d_ = sdS[wave][e][1];
        float s1 = 0.f, s2 = 0.f;
#pragma unroll
        for (int t = 0; t < 4; ++t) {
          int j = lane * 4 + t;
          float v = (j < 64) ? mg[(size_t)d_ * DD + j]
                             : mg[(size_t)s_ * DD + (j - 64)];
          v = fminf(fmaxf(v, 0.f), 10.f);
          featS[wave][e][j] = (_Float16)v;
          s1 += v; s2 += v * v;
        }
#pragma unroll
        for (int off = 16; off >= 1; off >>= 1) {
          s1 += __shfl_xor(s1, off, 32);
          s2 += __shfl_xor(s2, off, 32);
        }
        if (lane == 0) {
          float m = s1 * (1.f / 128.f);
          float var = s2 * (1.f / 128.f) - m * m;
          statS[wave][e][0] = m;
          statS[wave][e][1] = rsqrtf(var + 1e-5f);
        }
      }
    }
    __syncthreads();
    if (act) {
      float m = statS[wave][l15][0], rs = statS[wave][l15][1];
      v8f c0 = {}, c1 = {};
#pragma unroll
      for (int ck = 0; ck < 4; ++ck) {
        v16h a;
#pragma unroll
        for (int e = 0; e < 16; ++e) {
          int K = ck * 32 + kpat(e, half);
          float v = ((float)featS[wave][l15][K] - m) * rs * lng[K] + lnb[K];
          a[e] = (_Float16)v;
        }
        c0 = wmma16(a, bw[0][ck], c0);
        c1 = wmma16(a, bw[1][ck], c1);
      }
#pragma unroll
      for (int r = 0; r < 8; ++r) {       // hidden @ W2, cross-lane reduce
        float z0 = fmaxf(c0[r] + b1lo, 0.f);
        float z1 = fmaxf(c1[r] + b1hi, 0.f);
        float t = z0 * w2lo + z1 * w2hi;
        t += __shfl_xor(t, 1, 32);
        t += __shfl_xor(t, 2, 32);
        t += __shfl_xor(t, 4, 32);
        t += __shfl_xor(t, 8, 32);
        if (l15 == 0) {                   // lanes 0 (edge r) and 16 (edge r+8)
          int e_ = r + 8 * half;
          float cp = 1.f / (1.f + __expf(-(t + b2v)));
          float w = fminf(fmaxf(ewS[wave][e_], 0.1f), 2.f);
          float ph = fminf(fmaxf(0.3f * cp * w, -MAXPH), MAXPH);
          csS[wave][e_][0] = __cosf(ph);
          csS[wave][e_][1] = __sinf(ph);
        }
      }
    }
    __syncthreads();
    if (act) {
#pragma unroll 1
      for (int e = 0; e < 16; ++e) {      // rotate + scatter-add (L2 atomics)
        size_t sb = (size_t)sdS[wave][e][0] * DD;
        size_t db = (size_t)sdS[wave][e][1] * DD;
        float ce = csS[wave][e][0], se = csS[wave][e][1];
#pragma unroll
        for (int t = 0; t < 2; ++t) {
          int col = lane + t * 32;
          float p = P[sb + col], q = Q[sb + col];
          atomicAdd(&outR[db + col], ce * p - se * q);
          atomicAdd(&outI[db + col], se * p + ce * q);
        }
      }
    }
    __syncthreads();
  }
}

// ---------------------------------------------------------------------------
// U_small chain (tiny; VALU)
// ---------------------------------------------------------------------------
__global__ void u_deg_kernel(const int* __restrict__ src, const int* __restrict__ dst,
                             float* __restrict__ degf) {
  for (int e = blockIdx.x * blockDim.x + threadIdx.x; e < EE; e += gridDim.x * blockDim.x) {
    int s = src[e], d = dst[e];
    if ((unsigned)s < 64u && (unsigned)d < 64u) atomicAdd(&degf[d], 1.0f);
  }
}
__global__ void u_dinv_kernel(const float* __restrict__ degf, float* __restrict__ dinv) {
  int i = threadIdx.x;
  if (i < 64) dinv[i] = rsqrtf(degf[i] + 1.0f);
}
__global__ void u_edge1_kernel(const int* __restrict__ src, const int* __restrict__ dst,
                               const float* __restrict__ dinv, float* __restrict__ S1) {
  for (int e = blockIdx.x * blockDim.x + threadIdx.x; e < EE; e += gridDim.x * blockDim.x) {
    int s = src[e], d = dst[e];
    if ((unsigned)s < 64u && (unsigned)d < 64u) atomicAdd(&S1[d], dinv[s] * dinv[d]);
  }
}
__global__ void __launch_bounds__(256)
u_xw2_kernel(const float* __restrict__ S1, const float* __restrict__ dinv,
             const float* __restrict__ g1w, const float* __restrict__ g1b,
             const float* __restrict__ g2w, float* __restrict__ xw2) {
  __shared__ float h1[64 * 64];
  int tid = threadIdx.x;
  for (int idx = tid; idx < 4096; idx += 256) {
    int i = idx >> 6, f = idx & 63;
    float t = S1[i] + dinv[i] * dinv[i];
    h1[idx] = fmaxf(g1w[f] * t + g1b[f], 0.f);   // gcn1 out + relu
  }
  __syncthreads();
  for (int idx = tid; idx < 4096; idx += 256) {
    int i = idx >> 6, f = idx & 63;
    float acc = 0.f;
    for (int k = 0; k < 64; ++k) acc += h1[i * 64 + k] * g2w[k * 64 + f];
    xw2[idx] = acc;
  }
}
__global__ void u_edge2_kernel(const int* __restrict__ src, const int* __restrict__ dst,
                               const float* __restrict__ dinv,
                               const float* __restrict__ xw2, float* __restrict__ O2) {
  for (int e = blockIdx.x * blockDim.x + threadIdx.x; e < EE; e += gridDim.x * blockDim.x) {
    int s = src[e], d = dst[e];
    if ((unsigned)s < 64u && (unsigned)d < 64u) {
      float w = dinv[s] * dinv[d];
      for (int f = 0; f < 64; ++f) atomicAdd(&O2[d * 64 + f], xw2[s * 64 + f] * w);
    }
  }
}
__global__ void __launch_bounds__(256)
u_head_kernel(const float* __restrict__ O2, const float* __restrict__ xw2,
              const float* __restrict__ dinv, const float* __restrict__ g2b,
              const float* __restrict__ aw1, const float* __restrict__ ab1,
              const float* __restrict__ aw2, const float* __restrict__ ab2,
              const float* __restrict__ aw3, const float* __restrict__ ab3,
              float* __restrict__ pbuf) {
  __shared__ float g[64], z1[64], z2[64];
  int tid = threadIdx.x;
  if (tid < 64) {
    float acc = 0.f;
    for (int i = 0; i < 64; ++i)
      acc += O2[i * 64 + tid] + xw2[i * 64 + tid] * dinv[i] * dinv[i];
    g[tid] = acc * (1.f / 64.f) + g2b[tid];       // global mean pool (+bias)
  }
  __syncthreads();
  if (tid < 64) {
    float acc = ab1[tid];
    for (int k = 0; k < 64; ++k) acc += g[k] * aw1[k * 64 + tid];
    z1[tid] = fmaxf(acc, 0.f);
  }
  __syncthreads();
  if (tid < 64) {
    float acc = ab2[tid];
    for (int k = 0; k < 64; ++k) acc += z1[k] * aw2[k * 64 + tid];
    z2[tid] = fmaxf(acc, 0.f);
  }
  __syncthreads();
  for (int j = tid; j < 4096; j += 256) {
    float acc = ab3[j];
    for (int k = 0; k < 64; ++k) acc += z2[k] * aw3[k * 4096 + j];
    pbuf[j] = tanhf(acc);
  }
}

// expm(A), A = 0.05*(p-p^T) + i*0.05*(p+p^T); scaling 2^-8 + Taylor-12 + 8 squarings
__global__ void __launch_bounds__(256)
u_expm_kernel(const float* __restrict__ pbuf, float* __restrict__ Ur,
              float* __restrict__ Ui) {
  __shared__ float Br[4096], Bi[4096], Tr[4096], Ti[4096];
  int tid = threadIdx.x;
  const float sc = 0.05f * (1.0f / 256.0f);
  for (int idx = tid; idx < 4096; idx += 256) {
    int i = idx >> 6, j = idx & 63;
    float pij = pbuf[idx], pji = pbuf[j * 64 + i];
    Br[idx] = sc * (pij - pji);
    Bi[idx] = sc * (pij + pji);
    Tr[idx] = (i == j) ? 1.f : 0.f;
    Ti[idx] = 0.f;
  }
  __syncthreads();
  for (int k = 12; k >= 1; --k) {          // Horner: T = I + (B*T)/k
    float mr[16], mi[16];
    float invk = 1.f / (float)k;
#pragma unroll
    for (int t = 0; t < 16; ++t) {
      int idx = tid + t * 256;
      int i = idx >> 6, j = idx & 63;
      float ar = 0.f, ai = 0.f;
      for (int q = 0; q < 64; ++q) {
        float br = Br[i * 64 + q], bi = Bi[i * 64 + q];
        float tr = Tr[q * 64 + j], ti = Ti[q * 64 + j];
        ar += br * tr - bi * ti;
        ai += br * ti + bi * tr;
      }
      mr[t] = ar * invk + ((i == j) ? 1.f : 0.f);
      mi[t] = ai * invk;
    }
    __syncthreads();
#pragma unroll
    for (int t = 0; t < 16; ++t) { int idx = tid + t * 256; Tr[idx] = mr[t]; Ti[idx] = mi[t]; }
    __syncthreads();
  }
  for (int sq = 0; sq < 8; ++sq) {         // T = T*T
    float mr[16], mi[16];
#pragma unroll
    for (int t = 0; t < 16; ++t) {
      int idx = tid + t * 256;
      int i = idx >> 6, j = idx & 63;
      float ar = 0.f, ai = 0.f;
      for (int q = 0; q < 64; ++q) {
        float x0 = Tr[i * 64 + q], x1 = Ti[i * 64 + q];
        float y0 = Tr[q * 64 + j], y1 = Ti[q * 64 + j];
        ar += x0 * y0 - x1 * y1;
        ai += x0 * y1 + x1 * y0;
      }
      mr[t] = ar; mi[t] = ai;
    }
    __syncthreads();
#pragma unroll
    for (int t = 0; t < 16; ++t) { int idx = tid + t * 256; Tr[idx] = mr[t]; Ti[idx] = mi[t]; }
    __syncthreads();
  }
  for (int idx = tid; idx < 4096; idx += 256) { Ur[idx] = Tr[idx]; Ui[idx] = Ti[idx]; }
}

// ---------------------------------------------------------------------------
// Finalize: out = 0.5*aggr + 0.5*x_real ; out_imag = 0.5*aggr_imag
// ---------------------------------------------------------------------------
__global__ void finalize_kernel(const float* __restrict__ xr,
                                float* __restrict__ outR, float* __restrict__ outI) {
  const size_t total = (size_t)NN * DD;
  for (size_t i = blockIdx.x * (size_t)blockDim.x + threadIdx.x; i < total;
       i += (size_t)gridDim.x * blockDim.x) {
    outR[i] = 0.5f * outR[i] + 0.5f * xr[i];
    outI[i] = 0.5f * outI[i];
  }
}

// ---------------------------------------------------------------------------
extern "C" void kernel_launch(void* const* d_in, const int* in_sizes, int n_in,
                              void* d_out, int out_size, void* d_ws, size_t ws_size,
                              hipStream_t stream) {
  const float* x   = (const float*)d_in[0];
  const int*   ei  = (const int*)d_in[1];
  const float* ew  = (const float*)d_in[2];
  const float* Wr  = (const float*)d_in[3];
  // d_in[4] lin_imag_w unused: x_imag == 0
  const float* lng = (const float*)d_in[5];
  const float* lnb = (const float*)d_in[6];
  const float* W1  = (const float*)d_in[7];
  const float* b1  = (const float*)d_in[8];
  const float* W2  = (const float*)d_in[9];
  const float* b2  = (const float*)d_in[10];
  const float* g1w = (const float*)d_in[11];
  const float* g1b = (const float*)d_in[12];
  const float* g2w = (const float*)d_in[13];
  const float* g2b = (const float*)d_in[14];
  const float* aw1 = (const float*)d_in[15];
  const float* ab1 = (const float*)d_in[16];
  const float* aw2 = (const float*)d_in[17];
  const float* ab2 = (const float*)d_in[18];
  const float* aw3 = (const float*)d_in[19];
  const float* ab3 = (const float*)d_in[20];
  const int* srcI = ei;
  const int* dstI = ei + EE;

  float* ws = (float*)d_ws;
  const size_t NW = (size_t)NN * DD;
  float* xr = ws;
  float* h  = ws + NW;
  float* mg = ws + 2 * NW;
  float* P  = ws + 3 * NW;
  float* Q  = ws + 4 * NW;
  float* small = ws + 5 * NW;
  float* degf = small;                       // 64
  float* dinv = small + 64;                  // 64
  float* S1   = small + 128;                 // 64
  float* xw2  = small + 192;                 // 4096
  float* O2   = small + 192 + 4096;          // 4096
  float* pbuf = small + 192 + 2 * 4096;      // 4096
  float* Ur   = small + 192 + 3 * 4096;      // 4096
  float* Ui   = small + 192 + 4 * 4096;      // 4096

  float* outR = (float*)d_out;
  float* outI = outR + NW;

  hipMemsetAsync(d_out, 0, (size_t)out_size * sizeof(float), stream);
  hipMemsetAsync(small, 0, (size_t)(192 + 5 * 4096) * sizeof(float), stream);

  // node pipeline (independent of U chain)
  ln_h_mag_kernel<<<391, 256, 0, stream>>>(x, Wr, xr, h, mg);

  // U = expm(A) chain
  u_deg_kernel<<<512, 256, 0, stream>>>(srcI, dstI, degf);
  u_dinv_kernel<<<1, 64, 0, stream>>>(degf, dinv);
  u_edge1_kernel<<<512, 256, 0, stream>>>(srcI, dstI, dinv, S1);
  u_xw2_kernel<<<1, 256, 0, stream>>>(S1, dinv, g1w, g1b, g2w, xw2);
  u_edge2_kernel<<<512, 256, 0, stream>>>(srcI, dstI, dinv, xw2, O2);
  u_head_kernel<<<1, 256, 0, stream>>>(O2, xw2, dinv, g2b, aw1, ab1, aw2, ab2, aw3, ab3, pbuf);
  u_expm_kernel<<<1, 256, 0, stream>>>(pbuf, Ur, Ui);

  // P/Q projection, then the dominant edge pass
  pq_kernel<<<391, 256, 0, stream>>>(h, Ur, Ui, P, Q);
  const int nTiles = ETOT / 16;              // 53125, exact
  const int eBlocks = 512;
  const int nIter = (nTiles + eBlocks * 8 - 1) / (eBlocks * 8);
  edge_kernel<<<eBlocks, 256, 0, stream>>>(srcI, dstI, ew, mg, P, Q, lng, lnb,
                                           W1, b1, W2, b2, outR, outI, nIter, nTiles);
  finalize_kernel<<<1024, 256, 0, stream>>>(xr, outR, outI);
}
